// OCR_81492709474531
// MI455X (gfx1250) — matrix operand order repaired and checked by reading, not compiled, not run on hardware
//
#include <hip/hip_runtime.h>

// ---------------------------------------------------------------------------
// OCR head for MI455X (gfx1250): all GEMMs on v_wmma_f32_16x16x32_bf16 with
// f32 accumulation, double-buffered LDS staging on the CDNA5 async-to-LDS
// path (ASYNCcnt), one barrier per K-step, per-batch pipeline sized to stay
// resident in the 192 MB L2.
// ---------------------------------------------------------------------------

typedef __bf16 bf16_t;
typedef __attribute__((ext_vector_type(16))) __bf16 v16bf;
typedef __attribute__((ext_vector_type(8)))  __bf16 v8bf;
typedef __attribute__((ext_vector_type(4)))  __bf16 v4bf;
typedef __attribute__((ext_vector_type(8)))  float  v8f;
typedef __attribute__((ext_vector_type(4)))  int    v4i;

#define AS1 __attribute__((address_space(1)))
#define AS3 __attribute__((address_space(3)))

#if __has_builtin(__builtin_amdgcn_global_load_async_to_lds_b128)
#define HAVE_ASYNC_LDS 1
#else
#define HAVE_ASYNC_LDS 0
#endif

// 16-byte global -> LDS copy. Async path: memory->LDS directly (no VGPR
// round-trip), tracked by ASYNCcnt. Fallback: regular load + ds_store.
__device__ __forceinline__ void cp16_g2l(const bf16_t* g, bf16_t* l) {
#if HAVE_ASYNC_LDS
  __builtin_amdgcn_global_load_async_to_lds_b128(
      (AS1 v4i*)(AS1 void*)const_cast<bf16_t*>(g),
      (AS3 v4i*)(AS3 void*)l,
      0, 0);
#else
  *(uint4*)l = *(const uint4*)g;
#endif
}

__device__ __forceinline__ void wait_async_lds() {
#if HAVE_ASYNC_LDS
#if __has_builtin(__builtin_amdgcn_s_wait_asynccnt)
  __builtin_amdgcn_s_wait_asynccnt(0);
#else
  asm volatile("s_wait_asynccnt 0x0" ::: "memory");
#endif
#endif
}

// ---------------------------------------------------------------------------
// Tiled WMMA GEMM:  C[M,N] = act( scale * (A @ B) + bias )
//   A: [M,K] row-major (ATRANS=0) or [K,M] row-major (ATRANS=1)
//   B: [K,N] row-major (BTRANS=0; rows kk >= Ksplit read from B2[kk-Ksplit])
//      or [N,K] row-major (BTRANS=1)
//   EP: 0 = store f32, 1 = store bf16, 2 = atomicAdd f32 (split-K partials)
// Block: 256 thr = 8 wave32; tile 128x128x32; wave tile 32x64 (2x4 WMMA).
// Double-buffered LDS (40 KB of the 320 KB WGP pool), 1 barrier per K-step.
// All problem dims here are multiples of 128/32 -> no edge handling.
// ---------------------------------------------------------------------------
template<bool ATRANS, bool BTRANS, int EP, bool BIAS, bool RELU>
__global__ __launch_bounds__(256) void gemm_wmma_bf16(
    const bf16_t* __restrict__ A, int lda,
    const bf16_t* __restrict__ B1, const bf16_t* __restrict__ B2, int ldb, int Ksplit,
    const float* __restrict__ bias, float scale,
    float* __restrict__ outF, bf16_t* __restrict__ outB, int ldo,
    int M, int N, int K, int Kchunk)
{
  constexpr int BM = 128, BN = 128, BK = 32, S = 40; // S: padded LDS stride (bf16)
  __shared__ bf16_t As[2 * BM * S];   // As[buf][m][k]
  __shared__ bf16_t Bs[2 * BN * S];   // Bs[buf][n][k] (K-contiguous frag reads)
  const int tid  = threadIdx.x;
  const int lane = tid & 31, wave = tid >> 5;
  const int m0 = blockIdx.y * BM, n0 = blockIdx.x * BN;
  const int wm = (wave & 3) * 32, wn = (wave >> 2) * 64;
  v8f acc[2][4] = {};

  const int kbeg = blockIdx.z * Kchunk;
  int kend = kbeg + Kchunk; if (kend > K) kend = K;
  const int nsteps = (kend - kbeg) / BK;

  // Stage one 128x32 A tile + 128(n)x32(k) B tile into buffer `buf`.
  auto stage = [&](int k0, int buf) {
    bf16_t* Ab = As + buf * (BM * S);
    bf16_t* Bb = Bs + buf * (BN * S);
    if (!ATRANS) {                  // contiguous: async memory->LDS
      #pragma unroll
      for (int i = 0; i < 2; ++i) {
        int c = tid + i * 256;
        int r = c >> 2, k8 = (c & 3) * 8;
        cp16_g2l(A + (size_t)(m0 + r) * lda + (k0 + k8), Ab + r * S + k8);
      }
    } else {                        // A stored [K][M]: transpose in LDS store
      #pragma unroll
      for (int i = 0; i < 2; ++i) {
        int c = tid + i * 256;
        int kk = c >> 4, m8 = (c & 15) * 8;
        uint4 d = *(const uint4*)(A + (size_t)(k0 + kk) * lda + (m0 + m8));
        const bf16_t* e = (const bf16_t*)&d;
        #pragma unroll
        for (int j = 0; j < 8; ++j) Ab[(m8 + j) * S + kk] = e[j];
      }
    }
    if (BTRANS) {                   // B stored [N][K]: async straight copy
      #pragma unroll
      for (int i = 0; i < 2; ++i) {
        int c = tid + i * 256;
        int nn = c >> 2, k8 = (c & 3) * 8;
        cp16_g2l(B1 + (size_t)(n0 + nn) * ldb + (k0 + k8), Bb + nn * S + k8);
      }
    } else {                        // B stored [K][N] (split at Ksplit): transpose
      #pragma unroll
      for (int i = 0; i < 2; ++i) {
        int c = tid + i * 256;
        int kk = c >> 4, n8 = (c & 15) * 8;
        int kg = k0 + kk;
        const bf16_t* Bp = (kg < Ksplit) ? (B1 + (size_t)kg * ldb)
                                         : (B2 + (size_t)(kg - Ksplit) * ldb);
        uint4 d = *(const uint4*)(Bp + n0 + n8);
        const bf16_t* e = (const bf16_t*)&d;
        #pragma unroll
        for (int j = 0; j < 8; ++j) Bb[(n8 + j) * S + kk] = e[j];
      }
    }
  };

  stage(kbeg, 0);                   // prologue fill of buffer 0

  for (int s = 0; s < nsteps; ++s) {
    const int cur = s & 1;
    wait_async_lds();               // our async writes into buf `cur` landed
    __syncthreads();                // everyone's writes landed; prev reads done

    // ---- fragments from buf `cur` per ISA VGPR layouts ----
    union Frag { v16bf v; v8bf h[2]; };
    const bf16_t* Ac = As + cur * (BM * S);
    const bf16_t* Bc = Bs + cur * (BN * S);
    const int lm  = lane & 15;
    const int kbA = (lane >> 4) * 8;   // A: lanes 16-31 hold K=8..15 / 24..31
    const int kbB = (lane >> 4) * 8;   // B: lanes 16-31 hold K=8..15 / 24..31
    Frag a[2], bfr[4];
    #pragma unroll
    for (int mi = 0; mi < 2; ++mi) {
      const bf16_t* pA = Ac + (wm + mi * 16 + lm) * S;
      a[mi].h[0] = *(const v8bf*)(pA + kbA);
      a[mi].h[1] = *(const v8bf*)(pA + kbA + 16);
    }
    #pragma unroll
    for (int ni = 0; ni < 4; ++ni) {
      const bf16_t* pB = Bc + (wn + ni * 16 + lm) * S + kbB;
      bfr[ni].h[0] = *(const v8bf*)(pB);
      bfr[ni].h[1] = *(const v8bf*)(pB + 16);
    }

    // Prefetch next tile into the other buffer while the WMMAs run.
    if (s + 1 < nsteps) stage(kbeg + (s + 1) * BK, cur ^ 1);

    #pragma unroll
    for (int mi = 0; mi < 2; ++mi)
      #pragma unroll
      for (int ni = 0; ni < 4; ++ni)
        acc[mi][ni] = __builtin_amdgcn_wmma_f32_16x16x32_bf16(
            false, a[mi].v, false, bfr[ni].v, (short)0, acc[mi][ni], false, false);
  }

  // ---- epilogue: C/D layout -> lane (lm = N), VGPR r -> M = r (+8 hi half) ----
  const int lm  = lane & 15;
  const int mhi = (lane >> 4) * 8;
  #pragma unroll
  for (int mi = 0; mi < 2; ++mi) {
    #pragma unroll
    for (int ni = 0; ni < 4; ++ni) {
      int col = n0 + wn + ni * 16 + lm;
      #pragma unroll
      for (int r = 0; r < 8; ++r) {
        int row = m0 + wm + mi * 16 + mhi + r;
        float v = acc[mi][ni][r] * scale;
        if (BIAS) v += bias[row];
        if (RELU) v = fmaxf(v, 0.0f);
        size_t idx = (size_t)row * ldo + col;
        if (EP == 0)      outF[idx] = v;
        else if (EP == 1) outB[idx] = (bf16_t)v;
        else              atomicAdd(&outF[idx], v);
      }
    }
  }
}

// ---------------------------------------------------------------------------
// Row stats for spatial softmax: max + 1/sum(exp) over N per (c) row.
// ---------------------------------------------------------------------------
__global__ void rowstats_kernel(const float* __restrict__ x, float* __restrict__ mx,
                                float* __restrict__ rz, int N) {
  __shared__ float red[256];
  const float* p = x + (size_t)blockIdx.x * N;
  float m = -1e30f;
  for (int i = threadIdx.x; i < N; i += 256) m = fmaxf(m, p[i]);
  red[threadIdx.x] = m; __syncthreads();
  for (int s = 128; s > 0; s >>= 1) {
    if (threadIdx.x < s) red[threadIdx.x] = fmaxf(red[threadIdx.x], red[threadIdx.x + s]);
    __syncthreads();
  }
  m = red[0]; __syncthreads();
  float sum = 0.0f;
  for (int i = threadIdx.x; i < N; i += 256) sum += __expf(p[i] - m);
  red[threadIdx.x] = sum; __syncthreads();
  for (int s = 128; s > 0; s >>= 1) {
    if (threadIdx.x < s) red[threadIdx.x] += red[threadIdx.x + s];
    __syncthreads();
  }
  if (threadIdx.x == 0) { mx[blockIdx.x] = m; rz[blockIdx.x] = 1.0f / red[0]; }
}

// Elementwise: xb = bf16(x), pw = bf16(softmax_row(x))   (4 elems/thread)
__global__ void make_xb_pw_kernel(const float* __restrict__ x, const float* __restrict__ mx,
                                  const float* __restrict__ rz, bf16_t* __restrict__ xb,
                                  bf16_t* __restrict__ pw, int N) {
  int i = (blockIdx.x * 256 + threadIdx.x) * 4;
  int row = i / N;
  float m = mx[row], z = rz[row];
  float4 v = *(const float4*)(x + i);
  float f[4] = {v.x, v.y, v.z, v.w};
  v4bf xq, pq;
  #pragma unroll
  for (int j = 0; j < 4; ++j) {
    xq[j] = (bf16_t)f[j];
    pq[j] = (bf16_t)(__expf(f[j] - m) * z);
  }
  *(v4bf*)(xb + i) = xq;
  *(v4bf*)(pw + i) = pq;
}

// Softmax over K=512 attention logits per spatial row; emit bf16 weights.
__global__ void softmax512_kernel(const float* __restrict__ s, bf16_t* __restrict__ o) {
  __shared__ float red[256];
  const float* p = s + (size_t)blockIdx.x * 512;
  float a = p[threadIdx.x], b = p[threadIdx.x + 256];
  red[threadIdx.x] = fmaxf(a, b); __syncthreads();
  for (int st = 128; st > 0; st >>= 1) {
    if (threadIdx.x < st) red[threadIdx.x] = fmaxf(red[threadIdx.x], red[threadIdx.x + st]);
    __syncthreads();
  }
  float m = red[0]; __syncthreads();
  float ea = __expf(a - m), eb = __expf(b - m);
  red[threadIdx.x] = ea + eb; __syncthreads();
  for (int st = 128; st > 0; st >>= 1) {
    if (threadIdx.x < st) red[threadIdx.x] += red[threadIdx.x + st];
    __syncthreads();
  }
  float inv = 1.0f / red[0];
  bf16_t* q = o + (size_t)blockIdx.x * 512;
  q[threadIdx.x]       = (bf16_t)(ea * inv);
  q[threadIdx.x + 256] = (bf16_t)(eb * inv);
}

__global__ void f2b_kernel(const float* __restrict__ in, bf16_t* __restrict__ o, int n) {
  int i = blockIdx.x * 256 + threadIdx.x;
  if (i < n) o[i] = (bf16_t)in[i];
}

__global__ void zerof_kernel(float* __restrict__ o, int n) {
  int i = blockIdx.x * 256 + threadIdx.x;
  if (i < n) o[i] = 0.0f;
}

// ---------------------------------------------------------------------------
// Host orchestration
// ---------------------------------------------------------------------------
static inline size_t alignup(size_t x) { return (x + 255) & ~(size_t)255; }

extern "C" void kernel_launch(void* const* d_in, const int* in_sizes, int n_in,
                              void* d_out, int out_size, void* d_ws, size_t ws_size,
                              hipStream_t stream) {
  (void)in_sizes; (void)n_in; (void)out_size; (void)ws_size;
  constexpr int B = 4, C = 512, N = 16384, KC = 256;
  constexpr int BIG = 1 << 30;

  const float* feat = (const float*)d_in[0];           // [B,C,H,W] -> [B,C,N]
  // d_in[1] (prob) is dead in the reference forward.
  const float* wp1 = (const float*)d_in[2],  *bp1 = (const float*)d_in[3];
  const float* wp2 = (const float*)d_in[4],  *bp2 = (const float*)d_in[5];
  const float* wo1 = (const float*)d_in[6],  *bo1 = (const float*)d_in[7];
  const float* wo2 = (const float*)d_in[8],  *bo2 = (const float*)d_in[9];
  const float* wd  = (const float*)d_in[10], *bd  = (const float*)d_in[11];
  const float* wu  = (const float*)d_in[12], *bu  = (const float*)d_in[13];
  const float* wout= (const float*)d_in[14], *bout= (const float*)d_in[15];
  float* out = (float*)d_out;

  // ---- workspace bump allocator (~125 MB; per-batch buffers are reused) ----
  char* wsp = (char*)d_ws;
  auto alloc = [&](size_t bytes) { void* r = (void*)wsp; wsp += alignup(bytes); return r; };
  bf16_t* wp1b = (bf16_t*)alloc((size_t)KC * C  * 2);
  bf16_t* wp2b = (bf16_t*)alloc((size_t)KC * KC * 2);
  bf16_t* wo1b = (bf16_t*)alloc((size_t)KC * C  * 2);
  bf16_t* wo2b = (bf16_t*)alloc((size_t)KC * KC * 2);
  bf16_t* wdb  = (bf16_t*)alloc((size_t)KC * C  * 2);
  bf16_t* wub  = (bf16_t*)alloc((size_t)C  * KC * 2);
  bf16_t* woutb= (bf16_t*)alloc((size_t)C * 2 * C * 2);
  float*  mx   = (float*) alloc(C * 4);
  float*  rz   = (float*) alloc(C * 4);
  bf16_t* xb   = (bf16_t*)alloc((size_t)C * N * 2);     // x in bf16
  bf16_t* pw   = (bf16_t*)alloc((size_t)C * N * 2);     // spatial softmax weights
  float*  ctxf = (float*) alloc((size_t)C * C * 4);     // split-K partials
  bf16_t* ctxb = (bf16_t*)alloc((size_t)C * C * 2);
  bf16_t* t1   = (bf16_t*)alloc((size_t)KC * N * 2);
  bf16_t* qb   = (bf16_t*)alloc((size_t)KC * N * 2);
  bf16_t* t2   = (bf16_t*)alloc((size_t)KC * C * 2);
  bf16_t* kb   = (bf16_t*)alloc((size_t)KC * C * 2);
  bf16_t* vb   = (bf16_t*)alloc((size_t)KC * C * 2);
  float*  simf = (float*) alloc((size_t)N * C * 4);     // attention logits
  bf16_t* simb = (bf16_t*)alloc((size_t)N * C * 2);
  bf16_t* attn = (bf16_t*)alloc((size_t)KC * N * 2);
  bf16_t* up   = (bf16_t*)alloc((size_t)C * N * 2);

  const dim3 blk(256);

  // ---- weights f32 -> bf16 (once) ----
  f2b_kernel<<<(KC*C +255)/256, blk, 0, stream>>>(wp1, wp1b, KC*C);
  f2b_kernel<<<(KC*KC+255)/256, blk, 0, stream>>>(wp2, wp2b, KC*KC);
  f2b_kernel<<<(KC*C +255)/256, blk, 0, stream>>>(wo1, wo1b, KC*C);
  f2b_kernel<<<(KC*KC+255)/256, blk, 0, stream>>>(wo2, wo2b, KC*KC);
  f2b_kernel<<<(KC*C +255)/256, blk, 0, stream>>>(wd,  wdb,  KC*C);
  f2b_kernel<<<(C*KC +255)/256, blk, 0, stream>>>(wu,  wub,  C*KC);
  f2b_kernel<<<(C*2*C+255)/256, blk, 0, stream>>>(wout,woutb,C*2*C);

  for (int b = 0; b < B; ++b) {
    const float* xB = feat + (size_t)b * C * N;
    float* outB = out + (size_t)b * C * N;

    // spatial softmax stats + bf16 conversions
    rowstats_kernel<<<C, blk, 0, stream>>>(xB, mx, rz, N);
    make_xb_pw_kernel<<<(C * N) / 1024, blk, 0, stream>>>(xB, mx, rz, xb, pw, N);

    // context[C,C] = x @ pw^T   (K = 16384, split-K x16 with f32 atomics)
    zerof_kernel<<<(C * C) / 256, blk, 0, stream>>>(ctxf, C * C);
    gemm_wmma_bf16<false, true, 2, false, false>
        <<<dim3(C / 128, C / 128, 16), blk, 0, stream>>>(
        xb, N, pw, pw, N, BIG, nullptr, 1.0f, ctxf, nullptr, C, C, C, N, N / 16);
    f2b_kernel<<<(C * C) / 256, blk, 0, stream>>>(ctxf, ctxb, C * C);

    // q branch: t1 = relu(wp1 x + bp1); q = relu(wp2 t1 + bp2)
    gemm_wmma_bf16<false, false, 1, true, true>
        <<<dim3(N / 128, KC / 128, 1), blk, 0, stream>>>(
        wp1b, C, xb, xb, N, BIG, bp1, 1.0f, nullptr, t1, N, KC, N, C, C);
    gemm_wmma_bf16<false, false, 1, true, true>
        <<<dim3(N / 128, KC / 128, 1), blk, 0, stream>>>(
        wp2b, KC, t1, t1, N, BIG, bp2, 1.0f, nullptr, qb, N, KC, N, KC, KC);

    // k branch: t2 = relu(wo1 ctx); k = relu(wo2 t2);  v = relu(wd ctx)
    gemm_wmma_bf16<false, false, 1, true, true>
        <<<dim3(C / 128, KC / 128, 1), blk, 0, stream>>>(
        wo1b, C, ctxb, ctxb, C, BIG, bo1, 1.0f, nullptr, t2, C, KC, C, C, C);
    gemm_wmma_bf16<false, false, 1, true, true>
        <<<dim3(C / 128, KC / 128, 1), blk, 0, stream>>>(
        wo2b, KC, t2, t2, C, BIG, bo2, 1.0f, nullptr, kb, C, KC, C, KC, KC);
    gemm_wmma_bf16<false, false, 1, true, true>
        <<<dim3(C / 128, KC / 128, 1), blk, 0, stream>>>(
        wdb, C, ctxb, ctxb, C, BIG, bd, 1.0f, nullptr, vb, C, KC, C, C, C);

    // sim[N,C] = (q^T k) * 256^-0.5 ; softmax over K regions
    gemm_wmma_bf16<true, false, 0, false, false>
        <<<dim3(C / 128, N / 128, 1), blk, 0, stream>>>(
        qb, N, kb, kb, C, BIG, nullptr, 0.0625f, simf, nullptr, C, N, C, KC, KC);
    softmax512_kernel<<<N, blk, 0, stream>>>(simf, simb);

    // attn[KC,N] = v @ sim^T ; up = relu(wu attn + bu)
    gemm_wmma_bf16<false, true, 1, false, false>
        <<<dim3(N / 128, KC / 128, 1), blk, 0, stream>>>(
        vb, C, simb, simb, C, BIG, nullptr, 1.0f, nullptr, attn, N, KC, N, C, C);
    gemm_wmma_bf16<false, false, 1, true, true>
        <<<dim3(N / 128, C / 128, 1), blk, 0, stream>>>(
        wub, KC, attn, attn, N, BIG, bu, 1.0f, nullptr, up, N, C, N, KC, KC);

    // out = relu(wout @ concat([up, x]) + bout)  (split-K concat, no copy)
    gemm_wmma_bf16<false, false, 0, true, true>
        <<<dim3(N / 128, C / 128, 1), blk, 0, stream>>>(
        woutb, 2 * C, up, xb, N, C, bout, 1.0f, outB, nullptr, N, C, N, 2 * C, 2 * C);
  }
}
